// Mamba2MVBlock_39281770889619
// MI455X (gfx1250) — matrix operand order repaired and checked by reading, not compiled, not run on hardware
//
#include <hip/hip_runtime.h>

// ---------------- constants ----------------
static constexpr int Bc  = 4;
static constexpr int Lc  = 4096;
static constexpr int Dc  = 1024;
static constexpr int DIc = 2048;
static constexpr int Mc  = Bc * Lc;      // 16384 rows
static constexpr int DSc = 16;
static constexpr int KSYM = 9;

// epilogue modes
static constexpr int EPI_BF16  = 0;  // out_bf16[m*ldc + col_off + n] = v
static constexpr int EPI_SPLIT = 1;  // n<DI -> sigmoid->gate ; else -> val
static constexpr int EPI_GATE  = 2;  // out = v * gate[m,n]
static constexpr int EPI_RESID = 3;  // out_f32 = resid + v

// ---------------- bf16 helpers (storage = ushort) ----------------
__device__ __forceinline__ unsigned short f2bf(float f) {
  union { float f; unsigned int u; } v; v.f = f;
  unsigned int r = v.u + 0x7FFFu + ((v.u >> 16) & 1u);
  return (unsigned short)(r >> 16);
}
__device__ __forceinline__ float bf2f(unsigned short h) {
  union { unsigned int u; float f; } v; v.u = ((unsigned int)h) << 16;
  return v.f;
}

// ---------------- WMMA types ----------------
typedef __bf16 v16bf __attribute__((ext_vector_type(16)));
typedef float  v8f   __attribute__((ext_vector_type(8)));
union FragBF { v16bf bf; uint4 q[2]; };

// ---------------- CDNA5 async LDS copy (ASYNCcnt path) ----------------
__device__ __forceinline__ void async_ld_b128(unsigned lds_off, const void* gaddr) {
  asm volatile("global_load_async_to_lds_b128 %0, %1, off"
               :: "v"(lds_off), "v"(gaddr) : "memory");
}
__device__ __forceinline__ void wait_async0() {
  asm volatile("s_wait_asynccnt 0" ::: "memory");
}

// ---------------- weight convert ----------------
__global__ void cvt_f32_bf16(const float* __restrict__ in,
                             unsigned short* __restrict__ out, int n) {
  int i = blockIdx.x * 256 + threadIdx.x;
  if (i < n) out[i] = f2bf(in[i]);
}

// ---------------- RMSNorm (wave32 per row) ----------------
__global__ void rmsnorm_kernel(const float* __restrict__ x,
                               const float* __restrict__ w,
                               unsigned short* __restrict__ xn) {
  int wid  = threadIdx.x >> 5;
  int lane = threadIdx.x & 31;
  int m = blockIdx.x * 8 + wid;
  const float4* x4 = (const float4*)(x + (size_t)m * Dc);
  float4 v[8];
  float ss = 0.f;
#pragma unroll
  for (int i = 0; i < 8; ++i) {
    v[i] = x4[i * 32 + lane];
    ss += v[i].x * v[i].x + v[i].y * v[i].y + v[i].z * v[i].z + v[i].w * v[i].w;
  }
#pragma unroll
  for (int off = 16; off > 0; off >>= 1) ss += __shfl_xor(ss, off, 32);
  float scale = rsqrtf(ss * (1.0f / (float)Dc) + 1e-6f);
  ushort4* o4 = (ushort4*)(xn + (size_t)m * Dc);
#pragma unroll
  for (int i = 0; i < 8; ++i) {
    int idx4 = i * 32 + lane;
    const float4 wv = ((const float4*)w)[idx4];
    ushort4 o;
    o.x = f2bf(v[i].x * wv.x * scale);
    o.y = f2bf(v[i].y * wv.y * scale);
    o.z = f2bf(v[i].z * wv.z * scale);
    o.w = f2bf(v[i].w * wv.w * scale);
    o4[idx4] = o;
  }
}

// -------- generic bf16 WMMA GEMM: C = epi(A[M,K] @ W[N,K]^T + bias) ---------
// block = 256 threads = 8 waves (4 M x 2 N); block tile 128x128; wave 32x64.
// Double-buffered LDS fed by global_load_async_to_lds_b128.
static constexpr int BM = 128, BN = 128, BK = 32;

__global__ __launch_bounds__(256)
void gemm_bf16_wmma(const unsigned short* __restrict__ Abf,
                    const unsigned short* __restrict__ Wbf,
                    const float* __restrict__ bias,
                    int N, int K, int epi,
                    unsigned short* __restrict__ outb, int ldc, int col_off,
                    const unsigned short* __restrict__ gate,
                    float* __restrict__ outf,
                    const float* __restrict__ resid,
                    unsigned short* __restrict__ out2) {
  __shared__ __align__(16) unsigned short sA[2][BM][BK + 8];
  __shared__ __align__(16) unsigned short sB[2][BN][BK + 8];

  const int tid  = threadIdx.x;
  const int n0   = blockIdx.x * BN;
  const int m0   = blockIdx.y * BM;
  const int wave = tid >> 5, lane = tid & 31;
  const int waveM = wave & 3, waveN = wave >> 2;   // 4 x 2 wave grid
  const int lh = lane & 15;
  const bool hiHalf = lane >= 16;

  v8f acc[2][4];
#pragma unroll
  for (int u = 0; u < 2; ++u)
#pragma unroll
    for (int t = 0; t < 4; ++t)
      acc[u][t] = (v8f){0.f,0.f,0.f,0.f,0.f,0.f,0.f,0.f};

  // tile staging: 128 rows x 4 chunks(8 bf16) per matrix; each thread moves
  // rows r and r+64 of both A and B (4 async b128 per tile).
  const int r  = tid >> 2;      // 0..63
  const int kc = tid & 3;       // 0..3
  const unsigned short* gA0 = Abf + (size_t)(m0 + r) * K + kc * 8;
  const unsigned short* gA1 = gA0 + (size_t)64 * K;
  const unsigned short* gB0 = Wbf + (size_t)(n0 + r) * K + kc * 8;
  const unsigned short* gB1 = gB0 + (size_t)64 * K;
  unsigned ldsA0[2], ldsA1[2], ldsB0[2], ldsB1[2];
#pragma unroll
  for (int b = 0; b < 2; ++b) {
    ldsA0[b] = (unsigned)(size_t)&sA[b][r][kc * 8];
    ldsA1[b] = (unsigned)(size_t)&sA[b][r + 64][kc * 8];
    ldsB0[b] = (unsigned)(size_t)&sB[b][r][kc * 8];
    ldsB1[b] = (unsigned)(size_t)&sB[b][r + 64][kc * 8];
  }
  auto issue_tile = [&](int buf, int kt) {
    const size_t ko = (size_t)kt * BK;
    async_ld_b128(ldsA0[buf], gA0 + ko);
    async_ld_b128(ldsA1[buf], gA1 + ko);
    async_ld_b128(ldsB0[buf], gB0 + ko);
    async_ld_b128(ldsB1[buf], gB1 + ko);
  };

  const int kTiles = K / BK;
  issue_tile(0, 0);
  wait_async0();
  __syncthreads();

  for (int kt = 0; kt < kTiles; ++kt) {
    const int cur = kt & 1;
    if (kt + 1 < kTiles) issue_tile(cur ^ 1, kt + 1);

    // A fragments: 16x32 bf16; lanes 0-15 hold K {0..7,16..23}, lanes 16-31
    // hold K {8..15,24..31}; row = M.
    const int kb = hiHalf ? 8 : 0;
    FragBF a0, a1;
    a0.q[0] = *(const uint4*)&sA[cur][waveM * 32 + lh][kb];
    a0.q[1] = *(const uint4*)&sA[cur][waveM * 32 + lh][kb + 16];
    a1.q[0] = *(const uint4*)&sA[cur][waveM * 32 + 16 + lh][kb];
    a1.q[1] = *(const uint4*)&sA[cur][waveM * 32 + 16 + lh][kb + 16];

    const int kbb = hiHalf ? 8 : 0;
#pragma unroll
    for (int t = 0; t < 4; ++t) {
      // B fragment: 32x16; lane = column N, lanes 0-15 K {0..7,16..23}, 16-31 K {8..15,24..31}
      FragBF b;
      const int br = waveN * 64 + t * 16 + lh;
      b.q[0] = *(const uint4*)&sB[cur][br][kbb];
      b.q[1] = *(const uint4*)&sB[cur][br][kbb + 16];
      acc[0][t] = __builtin_amdgcn_wmma_f32_16x16x32_bf16(
          false, a0.bf, false, b.bf, (short)0, acc[0][t], false, false);
      acc[1][t] = __builtin_amdgcn_wmma_f32_16x16x32_bf16(
          false, a1.bf, false, b.bf, (short)0, acc[1][t], false, false);
    }
    wait_async0();
    __syncthreads();
  }

  // epilogue: C layout lanes 0-15 -> M=vgpr, lanes 16-31 -> M=vgpr+8 ; N = lane%16
#pragma unroll
  for (int u = 0; u < 2; ++u) {
#pragma unroll
    for (int t = 0; t < 4; ++t) {
      const int n = n0 + waveN * 64 + t * 16 + lh;
      const float bv = bias[n];
#pragma unroll
      for (int v = 0; v < 8; ++v) {
        const int m = m0 + waveM * 32 + u * 16 + (hiHalf ? v + 8 : v);
        const float val = acc[u][t][v] + bv;
        if (epi == EPI_BF16) {
          outb[(size_t)m * ldc + col_off + n] = f2bf(val);
        } else if (epi == EPI_SPLIT) {
          if (n < DIc)
            outb[(size_t)m * DIc + n] = f2bf(1.f / (1.f + __expf(-val)));
          else
            out2[(size_t)m * DIc + (n - DIc)] = f2bf(val);
        } else if (epi == EPI_GATE) {
          const float g = bf2f(gate[(size_t)m * DIc + n]);
          outb[(size_t)m * ldc + n] = f2bf(val * g);
        } else { // EPI_RESID
          outf[(size_t)m * Dc + n] = resid[(size_t)m * Dc + n] + val;
        }
      }
    }
  }
}

// ---------------- depthwise conv, 4 taps, pad (1,2) ----------------
__global__ void dwconv4_kernel(const unsigned short* __restrict__ val,
                               const float* __restrict__ kw,
                               const float* __restrict__ kb,
                               unsigned short* __restrict__ out) {
  int idx = blockIdx.x * 256 + threadIdx.x;     // Mc * DIc/4 threads
  int c4 = idx & 511;
  int m  = idx >> 9;
  int d  = c4 << 2;
  int l  = m & (Lc - 1);
  float a0 = kb[d], a1 = kb[d + 1], a2 = kb[d + 2], a3 = kb[d + 3];
#pragma unroll
  for (int j = 0; j < 4; ++j) {
    int lp = l + j - 1;
    if (lp < 0 || lp >= Lc) continue;
    const unsigned short* p = val + (size_t)(m + j - 1) * DIc + d;
    uint2 u = *(const uint2*)p;
    a0 += bf2f((unsigned short)(u.x & 0xffff))  * kw[(d + 0) * 4 + j];
    a1 += bf2f((unsigned short)(u.x >> 16))     * kw[(d + 1) * 4 + j];
    a2 += bf2f((unsigned short)(u.y & 0xffff))  * kw[(d + 2) * 4 + j];
    a3 += bf2f((unsigned short)(u.y >> 16))     * kw[(d + 3) * 4 + j];
  }
  unsigned short* op = out + (size_t)m * DIc + d;
  op[0] = f2bf(a0); op[1] = f2bf(a1); op[2] = f2bf(a2); op[3] = f2bf(a3);
}

// ---------------- depthwise conv, 9 taps, pad (4,4), + SiLU ----------------
__global__ void dwconv9_silu_kernel(const unsigned short* __restrict__ val,
                                    const float* __restrict__ kw,
                                    const float* __restrict__ kb,
                                    unsigned short* __restrict__ out) {
  int idx = blockIdx.x * 256 + threadIdx.x;
  int c4 = idx & 511;
  int m  = idx >> 9;
  int d  = c4 << 2;
  int l  = m & (Lc - 1);
  float a0 = kb[d], a1 = kb[d + 1], a2 = kb[d + 2], a3 = kb[d + 3];
#pragma unroll
  for (int j = 0; j < KSYM; ++j) {
    int lp = l + j - 4;
    if (lp < 0 || lp >= Lc) continue;
    const unsigned short* p = val + (size_t)(m + j - 4) * DIc + d;
    uint2 u = *(const uint2*)p;
    a0 += bf2f((unsigned short)(u.x & 0xffff))  * kw[(d + 0) * KSYM + j];
    a1 += bf2f((unsigned short)(u.x >> 16))     * kw[(d + 1) * KSYM + j];
    a2 += bf2f((unsigned short)(u.y & 0xffff))  * kw[(d + 2) * KSYM + j];
    a3 += bf2f((unsigned short)(u.y >> 16))     * kw[(d + 3) * KSYM + j];
  }
  a0 = a0 / (1.f + __expf(-a0));
  a1 = a1 / (1.f + __expf(-a1));
  a2 = a2 / (1.f + __expf(-a2));
  a3 = a3 / (1.f + __expf(-a3));
  unsigned short* op = out + (size_t)m * DIc + d;
  op[0] = f2bf(a0); op[1] = f2bf(a1); op[2] = f2bf(a2); op[3] = f2bf(a3);
}

// ---------------- low-rank contract: s[m,j] = sum_d a[m,d] * A[d,j] ----------
__global__ void lowrank_contract(const unsigned short* __restrict__ a,
                                 const float* __restrict__ Amat,
                                 float* __restrict__ s) {
  int j  = threadIdx.x & 15;
  int mi = threadIdx.x >> 4;
  int m  = blockIdx.x * 16 + mi;
  const unsigned short* row = a + (size_t)m * DIc;
  float acc = 0.f;
#pragma unroll 4
  for (int d = 0; d < DIc; ++d)
    acc += bf2f(row[d]) * Amat[d * DSc + j];
  s[(size_t)m * DSc + j] = acc;
}

// ---------------- low-rank expand: a2[m,d] = sum_s s[m,s] * Bm[s,d] ----------
__global__ void lowrank_expand(const float* __restrict__ s,
                               const float* __restrict__ Bm,
                               unsigned short* __restrict__ a2) {
  int idx = blockIdx.x * 256 + threadIdx.x;   // Mc * DIc
  int d = idx & (DIc - 1);
  int m = idx >> 11;
  const float* sr = s + (size_t)m * DSc;
  float acc = 0.f;
#pragma unroll
  for (int k = 0; k < DSc; ++k)
    acc += sr[k] * Bm[k * DIc + d];
  a2[(size_t)m * DIc + d] = f2bf(acc);
}

// ---------------- host launcher ----------------
extern "C" void kernel_launch(void* const* d_in, const int* in_sizes, int n_in,
                              void* d_out, int out_size, void* d_ws, size_t ws_size,
                              hipStream_t stream) {
  const float* x      = (const float*)d_in[0];
  const float* norm_w = (const float*)d_in[1];
  const float* in_w   = (const float*)d_in[2];
  const float* in_b   = (const float*)d_in[3];
  const float* dwa_k  = (const float*)d_in[4];
  const float* dwa_b  = (const float*)d_in[5];
  const float* Amat   = (const float*)d_in[6];
  const float* Bm     = (const float*)d_in[7];
  const float* sym_k  = (const float*)d_in[8];
  const float* sym_b  = (const float*)d_in[9];
  const float* pw_w   = (const float*)d_in[10];
  const float* pw_b   = (const float*)d_in[11];
  const float* ha_w   = (const float*)d_in[12];
  const float* ha_b   = (const float*)d_in[13];
  const float* hb_w   = (const float*)d_in[14];
  const float* hb_b   = (const float*)d_in[15];
  const float* fuse_w = (const float*)d_in[16];
  const float* fuse_b = (const float*)d_in[17];
  const float* out_w  = (const float*)d_in[18];
  const float* out_b  = (const float*)d_in[19];
  float* out = (float*)d_out;

  size_t off = 0;
  auto alloc = [&](size_t bytes) -> void* {
    void* p = (char*)d_ws + off;
    off += (bytes + 255) & ~(size_t)255;
    return p;
  };

  // bf16 weight copies
  unsigned short* w_in   = (unsigned short*)alloc((size_t)2 * DIc * Dc * 2);   // 4096x1024
  unsigned short* w_pw   = (unsigned short*)alloc((size_t)DIc * DIc * 2);
  unsigned short* w_ha   = (unsigned short*)alloc((size_t)(DIc / 2) * DIc * 2);
  unsigned short* w_hb   = (unsigned short*)alloc((size_t)(DIc / 2) * DIc * 2);
  unsigned short* w_fuse = (unsigned short*)alloc((size_t)DIc * DIc * 2);
  unsigned short* w_out  = (unsigned short*)alloc((size_t)Dc * DIc * 2);
  // activations
  unsigned short* xn   = (unsigned short*)alloc((size_t)Mc * Dc * 2);
  unsigned short* gate = (unsigned short*)alloc((size_t)Mc * DIc * 2);
  unsigned short* valb = (unsigned short*)alloc((size_t)Mc * DIc * 2);
  unsigned short* abuf = (unsigned short*)alloc((size_t)Mc * DIc * 2);
  unsigned short* a2   = (unsigned short*)alloc((size_t)Mc * DIc * 2);
  unsigned short* bbuf = (unsigned short*)alloc((size_t)Mc * DIc * 2);
  float*          sbuf = (float*)alloc((size_t)Mc * DSc * 4);
  // aliases (stream-ordered reuse)
  unsigned short* b2 = abuf;   // abuf dead after lowrank_contract
  unsigned short* y  = bbuf;   // bbuf dead after pw GEMM
  unsigned short* yg = valb;   // valb dead after dwconv9

  auto cvt = [&](const float* src, unsigned short* dst, int n) {
    cvt_f32_bf16<<<(n + 255) / 256, 256, 0, stream>>>(src, dst, n);
  };
  cvt(in_w,   w_in,   2 * DIc * Dc);
  cvt(pw_w,   w_pw,   DIc * DIc);
  cvt(ha_w,   w_ha,   (DIc / 2) * DIc);
  cvt(hb_w,   w_hb,   (DIc / 2) * DIc);
  cvt(fuse_w, w_fuse, DIc * DIc);
  cvt(out_w,  w_out,  Dc * DIc);

  // 1) RMSNorm -> bf16
  rmsnorm_kernel<<<Mc / 8, 256, 0, stream>>>(x, norm_w, xn);

  dim3 blk(256);
  // 2) in_proj: [M,1024] x [4096,1024]^T, split into sigmoid(gate) / val
  gemm_bf16_wmma<<<dim3(4096 / BN, Mc / BM), blk, 0, stream>>>(
      xn, w_in, in_b, 4096, Dc, EPI_SPLIT, gate, 0, 0, nullptr, nullptr, nullptr, valb);

  // 3) a-path: dwconv4 -> low-rank (DI->16->DI)
  dwconv4_kernel<<<(Mc * (DIc / 4)) / 256, blk, 0, stream>>>(valb, dwa_k, dwa_b, abuf);
  lowrank_contract<<<Mc / 16, blk, 0, stream>>>(abuf, Amat, sbuf);
  lowrank_expand<<<(Mc * DIc) / 256, blk, 0, stream>>>(sbuf, Bm, a2);

  // 4) b-path: dwconv9 + SiLU -> pw GEMM
  dwconv9_silu_kernel<<<(Mc * (DIc / 4)) / 256, blk, 0, stream>>>(valb, sym_k, sym_b, bbuf);
  gemm_bf16_wmma<<<dim3(DIc / BN, Mc / BM), blk, 0, stream>>>(
      bbuf, w_pw, pw_b, DIc, DIc, EPI_BF16, b2, DIc, 0, nullptr, nullptr, nullptr, nullptr);

  // 5) ha / hb into concat buffer y[:, :1024] and y[:, 1024:]
  gemm_bf16_wmma<<<dim3((DIc / 2) / BN, Mc / BM), blk, 0, stream>>>(
      a2, w_ha, ha_b, DIc / 2, DIc, EPI_BF16, y, DIc, 0, nullptr, nullptr, nullptr, nullptr);
  gemm_bf16_wmma<<<dim3((DIc / 2) / BN, Mc / BM), blk, 0, stream>>>(
      b2, w_hb, hb_b, DIc / 2, DIc, EPI_BF16, y, DIc, DIc / 2, nullptr, nullptr, nullptr, nullptr);

  // 6) fuse GEMM with gate multiply
  gemm_bf16_wmma<<<dim3(DIc / BN, Mc / BM), blk, 0, stream>>>(
      y, w_fuse, fuse_b, DIc, DIc, EPI_GATE, yg, DIc, 0, gate, nullptr, nullptr, nullptr);

  // 7) out GEMM with residual add, f32 store
  gemm_bf16_wmma<<<dim3(Dc / BN, Mc / BM), blk, 0, stream>>>(
      yg, w_out, out_b, Dc, DIc, EPI_RESID, nullptr, 0, 0, nullptr, out, x, nullptr);

  (void)in_sizes; (void)n_in; (void)out_size; (void)ws_size;
}
